// TemporalGNN_70635032150611
// MI455X (gfx1250) — matrix operand order repaired and checked by reading, not compiled, not run on hardware
//
#include <hip/hip_runtime.h>
#include <hip/hip_bf16.h>

// ---------- CDNA5 WMMA types ----------
typedef __bf16 v16bf __attribute__((ext_vector_type(16)));
typedef __bf16 v8bf  __attribute__((ext_vector_type(8)));
typedef float  v8f   __attribute__((ext_vector_type(8)));

__device__ __forceinline__ __bf16 f2bf(float f) { return (__bf16)f; }  // native cvt
__device__ __forceinline__ float bf2f(unsigned short b) {
    unsigned u = ((unsigned)b) << 16;
    return __builtin_bit_cast(float, u);
}
__device__ __forceinline__ unsigned pk2(float a, float b) {
    unsigned short lo = __builtin_bit_cast(unsigned short, (__bf16)a);
    unsigned short hi = __builtin_bit_cast(unsigned short, (__bf16)b);
    return (unsigned)lo | ((unsigned)hi << 16);
}

// =====================================================================
// Weight prep: W[batch][K][N] f32 -> Wt[batch][N][K] bf16 so WMMA
// B-fragments are two runs of 8 K values per lane, 16 apart (two 16B loads).
// =====================================================================
__global__ void wprep_kernel(const float* __restrict__ W, __bf16* __restrict__ Wt,
                             int K, int Ncols, size_t total)
{
    size_t idx = (size_t)blockIdx.x * blockDim.x + threadIdx.x;
    const size_t stride = (size_t)gridDim.x * blockDim.x;
    const size_t mat = (size_t)K * Ncols;
    for (; idx < total; idx += stride) {
        const size_t b = idx / mat;
        const size_t r = idx - b * mat;
        const int k = (int)(r / Ncols);
        const int n = (int)(r - (size_t)k * Ncols);
        Wt[b * mat + (size_t)n * K + k] = f2bf(W[idx]);
    }
}

// =====================================================================
// Fused GEMM: out_bf16[M x 256] = relu( (X1f/deg) @ W1 (+ X2bf @ W2) + bias )
//  - X1f: f32, staged via LDS (deg-scale + bf16 convert), row-clamped loads
//  - X2bf: bf16, A-fragments direct from global (no LDS, no barriers)
//  - W*t : bf16 pre-transposed [col][K], B-fragments direct from global
// Block: 256 thr = 8 waves; block tile 128x64; wave tile 16x64 (4 accs).
// =====================================================================
#define LDSX_S 40   // X-tile row stride in halfs (80B rows, 16B-aligned chunks)

__global__ void __launch_bounds__(256)
gemm_bf16_kernel(const float* __restrict__ X1f, const float* __restrict__ deg,
                 const __bf16* __restrict__ W1t, int K1,
                 const __bf16* __restrict__ X2bf, const __bf16* __restrict__ W2t, int K2,
                 const float* __restrict__ bias,
                 __bf16* __restrict__ out, int Mnodes)
{
    __shared__ __align__(16) __bf16 sX[128 * LDSX_S];

    const int tid     = threadIdx.x;
    const int wave    = tid >> 5;
    const int lane    = tid & 31;
    const int halfsel = lane >> 4;
    const int l15     = lane & 15;

    const int mBlock = blockIdx.x * 128;
    const int nBlock = blockIdx.y * 64;

    const int abase = halfsel << 3;   // A: K sub-base 0 or 8
    const int koff  = halfsel << 3;   // B: K sub-base 0 or 8
    const int arow  = mBlock + (wave << 4) + l15;            // A global row
    const int arowc = arow < Mnodes ? arow : (Mnodes - 1);   // clamped (OOB rows discarded)

    v8f acc[4] = {{}, {}, {}, {}};

    // X1 tile loader mapping: 2 threads/row, 16 floats each
    const int xrow  = tid >> 1;
    const int xc0   = (tid & 1) << 4;
    const int grow  = mBlock + xrow;
    const int growc = grow < Mnodes ? grow : (Mnodes - 1);   // clamped

    auto ldB = [&](const __bf16* __restrict__ Wt, int K, int col, int kt) -> v16bf {
        v8bf lo = *(const v8bf*)&Wt[(size_t)col * K + kt + koff];
        v8bf hi = *(const v8bf*)&Wt[(size_t)col * K + kt + 16 + koff];
        v16bf b;
        #pragma unroll
        for (int i = 0; i < 8; ++i) { b[i] = lo[i]; b[i + 8] = hi[i]; }
        return b;
    };

    // ---------------- pair 1: f32 X1 via LDS ----------------
    {
        float inv = 1.0f;
        if (deg) inv = 1.0f / fmaxf(deg[growc], 1.0f);

        for (int kt = 0; kt < K1; kt += 32) {
            #pragma unroll
            for (int i = 0; i < 4; ++i) {
                const float4 v = *(const float4*)&X1f[(size_t)growc * K1 + kt + xc0 + 4 * i];
                uint2 p;
                p.x = pk2(v.x * inv, v.y * inv);
                p.y = pk2(v.z * inv, v.w * inv);
                *(uint2*)&sX[xrow * LDSX_S + xc0 + 4 * i] = p;  // one 8B LDS store
            }
            __syncthreads();

            const int mrow = (wave << 4) + l15;
            v8bf alo = *(const v8bf*)&sX[mrow * LDSX_S + abase];
            v8bf ahi = *(const v8bf*)&sX[mrow * LDSX_S + 16 + abase];
            v16bf a;
            #pragma unroll
            for (int i = 0; i < 8; ++i) { a[i] = alo[i]; a[i + 8] = ahi[i]; }

            #pragma unroll
            for (int j = 0; j < 4; ++j) {
                v16bf b = ldB(W1t, K1, nBlock + l15 + 16 * j, kt);
                acc[j] = __builtin_amdgcn_wmma_f32_16x16x32_bf16(false, a, false, b,
                                                                 (short)0, acc[j],
                                                                 false, false);
            }
            __syncthreads();
        }
    }

    // ---------------- pair 2: bf16 X2, fully direct-from-global ----------------
    if (X2bf) {
        #pragma unroll 2
        for (int kt = 0; kt < K2; kt += 32) {
            v8bf alo = *(const v8bf*)&X2bf[(size_t)arowc * 256 + kt + abase];
            v8bf ahi = *(const v8bf*)&X2bf[(size_t)arowc * 256 + kt + 16 + abase];
            v16bf a;
            #pragma unroll
            for (int i = 0; i < 8; ++i) { a[i] = alo[i]; a[i + 8] = ahi[i]; }

            #pragma unroll
            for (int j = 0; j < 4; ++j) {
                v16bf b = ldB(W2t, K2, nBlock + l15 + 16 * j, kt);
                acc[j] = __builtin_amdgcn_wmma_f32_16x16x32_bf16(false, a, false, b,
                                                                 (short)0, acc[j],
                                                                 false, false);
            }
        }
    }

    // ---------------- epilogue: bias + relu -> bf16 ----------------
    const int gmBase = mBlock + (wave << 4) + (halfsel << 3);
    #pragma unroll
    for (int j = 0; j < 4; ++j) {
        const int c  = nBlock + l15 + 16 * j;
        const float bv = bias[c];
        #pragma unroll
        for (int r = 0; r < 8; ++r) {
            const int gm = gmBase + r;
            if (gm < Mnodes)
                out[(size_t)gm * 256 + c] = f2bf(fmaxf(acc[j][r] + bv, 0.0f));
        }
    }
}

// =====================================================================
// Edge gather (bf16 rows, 512B each) + f32 scatter-add + degree.
// 64 lanes/edge, 8B (4 bf16) per lane.
// =====================================================================
__global__ void __launch_bounds__(256)
edge_kernel(const __bf16* __restrict__ h,
            const long long* __restrict__ src, const long long* __restrict__ dst,
            float* __restrict__ agg, float* __restrict__ deg, int E)
{
    const int e = blockIdx.x * 4 + (threadIdx.x >> 6);
    if (e >= E) return;
    const int lane = threadIdx.x & 63;
    const long long s = src[e];
    const long long d = dst[e];

    const uint2 hv = *(const uint2*)(h + (size_t)s * 256 + lane * 4);
    float* a = &agg[(size_t)d * 256 + lane * 4];
    unsafeAtomicAdd(a + 0, bf2f((unsigned short)(hv.x & 0xFFFFu)));
    unsafeAtomicAdd(a + 1, bf2f((unsigned short)(hv.x >> 16)));
    unsafeAtomicAdd(a + 2, bf2f((unsigned short)(hv.y & 0xFFFFu)));
    unsafeAtomicAdd(a + 3, bf2f((unsigned short)(hv.y >> 16)));
    if (lane == 0) unsafeAtomicAdd(&deg[d], 1.0f);
}

// ---------- zero scratch (agg + deg contiguous) ----------
__global__ void zero_kernel(float* __restrict__ p, size_t n)
{
    size_t i = (size_t)blockIdx.x * blockDim.x + threadIdx.x;
    const size_t stride = (size_t)gridDim.x * blockDim.x;
    for (; i < n; i += stride) p[i] = 0.0f;
}

// =====================================================================
// Final projection 256 -> 2 + log_softmax.  h is bf16; W_post in LDS.
// =====================================================================
__global__ void __launch_bounds__(256)
post_kernel(const __bf16* __restrict__ h, const float* __restrict__ Wp,
            const float* __restrict__ bp, float* __restrict__ out, int M)
{
    __shared__ float sw[512];
    sw[threadIdx.x]       = Wp[threadIdx.x];
    sw[threadIdx.x + 256] = Wp[threadIdx.x + 256];
    __syncthreads();

    int i = blockIdx.x * blockDim.x + threadIdx.x;
    const int stride = gridDim.x * blockDim.x;
    for (; i < M; i += stride) {
        const unsigned* row = (const unsigned*)(h + (size_t)i * 256);
        float o0 = bp[0], o1 = bp[1];
        #pragma unroll 4
        for (int kk = 0; kk < 128; ++kk) {
            const unsigned w = row[kk];
            const float h0 = bf2f((unsigned short)(w & 0xFFFFu));
            const float h1 = bf2f((unsigned short)(w >> 16));
            o0 += h0 * sw[4 * kk]     + h1 * sw[4 * kk + 2];
            o1 += h0 * sw[4 * kk + 1] + h1 * sw[4 * kk + 3];
        }
        const float m   = fmaxf(o0, o1);
        const float lse = m + __logf(__expf(o0 - m) + __expf(o1 - m));
        out[(size_t)i * 2 + 0] = o0 - lse;
        out[(size_t)i * 2 + 1] = o1 - lse;
    }
}

// =====================================================================
extern "C" void kernel_launch(void* const* d_in, const int* in_sizes, int n_in,
                              void* d_out, int out_size, void* d_ws, size_t ws_size,
                              hipStream_t stream)
{
    const float*     x      = (const float*)d_in[0];
    const long long* ei     = (const long long*)d_in[1];
    const float*     W_pre  = (const float*)d_in[2];
    const float*     b_pre  = (const float*)d_in[3];
    const float*     W_l    = (const float*)d_in[4];
    const float*     b_l    = (const float*)d_in[5];
    const float*     W_r    = (const float*)d_in[6];
    const float*     W_post = (const float*)d_in[7];
    const float*     b_post = (const float*)d_in[8];

    const int INF = 128, H = 256, T = 4;
    const int N = in_sizes[0] / INF;
    const int E = in_sizes[1] / (T * 2);

    // -------- workspace layout (bf16 region first, then f32 region) --------
    __bf16* hA  = (__bf16*)d_ws;                     // N*H
    __bf16* hB  = hA + (size_t)N * H;                // N*H
    __bf16* WpT = hB + (size_t)N * H;                // 128*256
    __bf16* WlT = WpT + (size_t)INF * H;             // 4*256*256
    __bf16* WrT = WlT + (size_t)T * H * H;           // 4*256*256
    float*  agg = (float*)(WrT + (size_t)T * H * H); // N*H (16B-aligned)
    float*  deg = agg + (size_t)N * H;               // N

    // -------- weight prep (bf16, transposed [col][K]) --------
    wprep_kernel<<<128, 256, 0, stream>>>(W_pre, WpT, INF, H, (size_t)INF * H);
    wprep_kernel<<<512, 256, 0, stream>>>(W_l, WlT, H, H, (size_t)T * H * H);
    wprep_kernel<<<512, 256, 0, stream>>>(W_r, WrT, H, H, (size_t)T * H * H);

    const dim3 gblk(256);
    const dim3 ggrid((N + 127) / 128, H / 64);

    // h = relu(x @ W_pre + b_pre)   (bf16 output)
    gemm_bf16_kernel<<<ggrid, gblk, 0, stream>>>(x, nullptr, WpT, INF,
                                                 nullptr, nullptr, 0,
                                                 b_pre, hA, N);

    __bf16* cur = hA;
    __bf16* nxt = hB;
    const size_t zn = (size_t)N * H + N;   // agg + deg

    for (int count = 0; count < T; ++count) {
        const int t = T - 1 - count;
        const long long* src = ei + (size_t)t * 2 * E;
        const long long* dst = src + E;

        zero_kernel<<<2048, 256, 0, stream>>>(agg, zn);
        edge_kernel<<<(E + 3) / 4, 256, 0, stream>>>(cur, src, dst, agg, deg, E);
        // h = relu( (agg/deg) @ W_l + b_l + h @ W_r )
        gemm_bf16_kernel<<<ggrid, gblk, 0, stream>>>(agg, deg,
                                                     WlT + (size_t)count * H * H, H,
                                                     cur,
                                                     WrT + (size_t)count * H * H, H,
                                                     b_l + (size_t)count * H,
                                                     nxt, N);
        __bf16* tmp = cur; cur = nxt; nxt = tmp;
    }

    post_kernel<<<(N + 255) / 256, 256, 0, stream>>>(cur, W_post, b_post,
                                                     (float*)d_out, N);
}
